// BitLinear_52707838656621
// MI455X (gfx1250) — 8-bit matrix operand order repaired and checked by reading, compiled, not run on hardware
//
#include <hip/hip_runtime.h>
#include <hip/hip_bf16.h>

typedef __attribute__((ext_vector_type(8))) int v8i;
typedef __attribute__((ext_vector_type(4))) int v4i;
typedef __attribute__((ext_vector_type(4))) unsigned int v4u;

#define D_IN   4096
#define D_OUT  4096
#define M_TOT  8192
#define W_ELEMS (D_OUT * D_IN)
#define QB 127.0f
#define EPS 1e-8f

#if defined(__has_builtin)
#if __has_builtin(__builtin_amdgcn_tensor_load_to_lds) && \
    __has_builtin(__builtin_amdgcn_s_wait_tensorcnt)
#define HAVE_TDM 1
#endif
#endif


__global__ __launch_bounds__(256) void wabs_partial(const float* __restrict__ w,
                                                    float* __restrict__ partial) {
    __shared__ float red[256];
    const float4* w4 = (const float4*)w;
    const int t = threadIdx.x;
    const long base = (long)blockIdx.x * 16384;
    float s = 0.f;
#pragma unroll 4
    for (int i = 0; i < 64; ++i) {
        float4 v = w4[base + t + i * 256];
        s += fabsf(v.x) + fabsf(v.y) + fabsf(v.z) + fabsf(v.w);
    }
    red[t] = s;
    __syncthreads();
    for (int off = 128; off > 0; off >>= 1) {
        if (t < off) red[t] += red[t + off];
        __syncthreads();
    }
    if (t == 0) partial[blockIdx.x] = red[0];
}

__global__ __launch_bounds__(256) void wabs_final(const float* __restrict__ partial,
                                                  float* __restrict__ wscale) {
    __shared__ float red[256];
    const int t = threadIdx.x;
    red[t] = partial[t];
    __syncthreads();
    for (int off = 128; off > 0; off >>= 1) {
        if (t < off) red[t] += red[t + off];
        __syncthreads();
    }
    if (t == 0) wscale[0] = red[0] / (float)W_ELEMS + EPS;
}

__device__ __forceinline__ unsigned int pack4_clamp(float a, float b, float c, float d,
                                                    float k, float lim) {
    int q0 = (int)fminf(fmaxf(rintf(a * k), -lim), lim);
    int q1 = (int)fminf(fmaxf(rintf(b * k), -lim), lim);
    int q2 = (int)fminf(fmaxf(rintf(c * k), -lim), lim);
    int q3 = (int)fminf(fmaxf(rintf(d * k), -lim), lim);
    return ((unsigned int)(unsigned char)(signed char)q0) |
           ((unsigned int)(unsigned char)(signed char)q1 << 8) |
           ((unsigned int)(unsigned char)(signed char)q2 << 16) |
           ((unsigned int)(unsigned char)(signed char)q3 << 24);
}

__global__ __launch_bounds__(256) void wquant(const float* __restrict__ w,
                                              const float* __restrict__ wscale,
                                              unsigned int* __restrict__ wq) {
    const float inv = 1.0f / wscale[0];
    const long i = (long)blockIdx.x * 256 + threadIdx.x;
    const float4* w4 = (const float4*)w;
    float4 v = w4[i];
    wq[i] = pack4_clamp(v.x, v.y, v.z, v.w, inv, 1.0f);
}

__global__ __launch_bounds__(256) void aquant(const float* __restrict__ x,
                                              unsigned int* __restrict__ xq,
                                              float* __restrict__ xscale) {
    __shared__ float red[256];
    const int tok = blockIdx.x;
    const int t = threadIdx.x;
    const float4* x4 = (const float4*)(x + (long)tok * D_IN);
    float4 v[4];
    float m = 0.f;
#pragma unroll
    for (int i = 0; i < 4; ++i) {
        v[i] = x4[t + i * 256];
        m = fmaxf(m, fmaxf(fmaxf(fabsf(v[i].x), fabsf(v[i].y)),
                           fmaxf(fabsf(v[i].z), fabsf(v[i].w))));
    }
    red[t] = m;
    __syncthreads();
    for (int off = 128; off > 0; off >>= 1) {
        if (t < off) red[t] = fmaxf(red[t], red[t + off]);
        __syncthreads();
    }
    const float scale = fmaxf(red[0], EPS);
    if (t == 0) xscale[tok] = scale;
    const float k = QB / scale;
    unsigned int* row = xq + (long)tok * (D_IN / 4);
#pragma unroll
    for (int i = 0; i < 4; ++i)
        row[t + i * 256] = pack4_clamp(v[i].x, v[i].y, v[i].z, v[i].w, k, 127.0f);
}

#if HAVE_TDM
__device__ __forceinline__ void tdm_load_2d(unsigned lds_addr, const void* gaddr,
                                            unsigned tensor_d0, unsigned tensor_d1,
                                            unsigned tile_d0, unsigned tile_d1,
                                            unsigned stride0) {
    const unsigned long long ga = (unsigned long long)gaddr;
    v4u g0 = { 1u,
               lds_addr,
               (unsigned)ga,
               (unsigned)((ga >> 32) & 0x1FFFFFFu) | 0x80000000u };
    v8i g1 = { 0,
               (int)(tensor_d0 << 16),
               (int)((tensor_d0 >> 16) | (tensor_d1 << 16)),
               (int)((tensor_d1 >> 16) | (tile_d0 << 16)),
               (int)(tile_d1 & 0xFFFFu),
               (int)stride0,
               0, 0 };
    v4i z = { 0, 0, 0, 0 };
#if __clang_major__ >= 23
    v8i z8 = { 0, 0, 0, 0, 0, 0, 0, 0 };
    __builtin_amdgcn_tensor_load_to_lds(g0, g1, z, z, z8, 0);
#else
    __builtin_amdgcn_tensor_load_to_lds(g0, g1, z, z, 0);
#endif
}
#endif

__global__ __launch_bounds__(256) void gemm_wmma(const unsigned char* __restrict__ xq,
                                                 const unsigned char* __restrict__ wq,
                                                 const float* __restrict__ xscale,
                                                 const float* __restrict__ wscale,
                                                 float* __restrict__ out) {
    __shared__ __align__(16) unsigned char smem[2 * 16384];

    const int t = threadIdx.x;
    const int lane = t & 31;
    const int wid = t >> 5;
    const int waveM = wid >> 1;
    const int waveN = wid & 1;
    const int half = lane >> 4;
    const int l16 = lane & 15;

    const long gM = (long)blockIdx.y * 128;
    const long gN = (long)blockIdx.x * 128;

    v8i acc[2][4];
    const v8i vzero = {0, 0, 0, 0, 0, 0, 0, 0};
#pragma unroll
    for (int im = 0; im < 2; ++im)
#pragma unroll
        for (int jn = 0; jn < 4; ++jn) acc[im][jn] = vzero;

    const int NS = D_IN / 64;

#if HAVE_TDM
    const unsigned lds_base = (unsigned)(unsigned long long)(void*)&smem[0];
    if (wid == 0) {
        tdm_load_2d(lds_base,        xq + gM * D_IN, D_IN, M_TOT, 64, 128, D_IN);
        tdm_load_2d(lds_base + 8192, wq + gN * D_IN, D_IN, D_OUT, 64, 128, D_IN);
    }
#else
    const int4* xg = (const int4*)xq;
    const int4* wg = (const int4*)wq;
#endif

    for (int ks = 0; ks < NS; ++ks) {
        const int cur = ks & 1;
        const int k0 = ks * 64;

#if HAVE_TDM
        if (wid == 0) {
            if (ks + 1 < NS) {
                const unsigned dst = lds_base + (unsigned)((cur ^ 1) * 16384);
                tdm_load_2d(dst,        xq + gM * D_IN + (k0 + 64), D_IN, M_TOT, 64, 128, D_IN);
                tdm_load_2d(dst + 8192, wq + gN * D_IN + (k0 + 64), D_IN, D_OUT, 64, 128, D_IN);
                __builtin_amdgcn_s_wait_tensorcnt(2);
            } else {
                __builtin_amdgcn_s_wait_tensorcnt(0);
            }
        }
        __syncthreads();
#else
        __syncthreads();
#pragma unroll
        for (int c = t; c < 512; c += 256) {
            const int row = c >> 2, seg = c & 3;
            int4* sA4 = (int4*)(smem + cur * 16384);
            int4* sB4 = (int4*)(smem + cur * 16384 + 8192);
            sA4[c] = xg[(gM + row) * (D_IN / 16) + (k0 >> 4) + seg];
            sB4[c] = wg[(gN + row) * (D_IN / 16) + (k0 >> 4) + seg];
        }
        __syncthreads();
#endif

        const int* lsA = (const int*)(smem + cur * 16384);
        const int* lsB = (const int*)(smem + cur * 16384 + 8192);

        v8i afr[2];
#pragma unroll
        for (int im = 0; im < 2; ++im) {
            const int rowi = (waveM * 32 + im * 16 + l16) * 16;
#pragma unroll
            for (int g = 0; g < 4; ++g) {
                const int kb = (g * 16 + half * 8) >> 2;
                afr[im][g * 2 + 0] = lsA[rowi + kb + 0];
                afr[im][g * 2 + 1] = lsA[rowi + kb + 1];
            }
        }
        v8i bfr[4];
#pragma unroll
        for (int jn = 0; jn < 4; ++jn) {
            const int n = waveN * 64 + jn * 16 + l16;
            const int* bp = lsB + n * 16 + half * 2;
#pragma unroll
            for (int q = 0; q < 8; ++q) bfr[jn][q] = bp[(q >> 1) * 4 + (q & 1)];
        }

#pragma unroll
        for (int im = 0; im < 2; ++im)
#pragma unroll
            for (int jn = 0; jn < 4; ++jn)
                acc[im][jn] = __builtin_amdgcn_wmma_i32_16x16x64_iu8(
                     true, afr[im],  true, bfr[jn],
                    acc[im][jn],  false,  false);

#if HAVE_TDM
        __syncthreads();
#endif
    }

    const float ws127 = wscale[0] * (1.0f / QB);
#pragma unroll
    for (int im = 0; im < 2; ++im) {
#pragma unroll
        for (int r = 0; r < 8; ++r) {
            const long row = gM + waveM * 32 + im * 16 + half * 8 + r;
            const float sc = xscale[row] * ws127;
#pragma unroll
            for (int jn = 0; jn < 4; ++jn) {
                const long col = gN + waveN * 64 + jn * 16 + l16;
                out[row * D_OUT + col] = (float)acc[im][jn][r] * sc;
            }
        }
    }
}

extern "C" void kernel_launch(void* const* d_in, const int* in_sizes, int n_in,
                              void* d_out, int out_size, void* d_ws, size_t ws_size,
                              hipStream_t stream) {
    (void)in_sizes; (void)n_in; (void)out_size; (void)ws_size;
    const float* x = (const float*)d_in[0];
    const float* w = (const float*)d_in[1];
    float* out = (float*)d_out;

    char* ws = (char*)d_ws;
    float* partial = (float*)ws;
    float* wscale  = (float*)(ws + 4096);
    float* xscalep = (float*)(ws + 8192);
    unsigned char* wq = (unsigned char*)(ws + 65536);
    unsigned char* xq = wq + (size_t)W_ELEMS;

    wabs_partial<<<256, 256, 0, stream>>>(w, partial);
    wabs_final<<<1, 256, 0, stream>>>(partial, wscale);
    wquant<<<W_ELEMS / (256 * 4), 256, 0, stream>>>(w, wscale, (unsigned int*)wq);
    aquant<<<M_TOT, 256, 0, stream>>>(x, (unsigned int*)xq, xscalep);
    gemm_wmma<<<dim3(D_OUT / 128, M_TOT / 128), 256, 0, stream>>>(
        xq, wq, xscalep, wscale, out);
}
